// BiLSTM_CRF_changed_44281112821764
// MI455X (gfx1250) — matrix operand order repaired and checked by reading, not compiled, not run on hardware
//
#include <hip/hip_runtime.h>
#include <hip/hip_bf16.h>
#include <math.h>

#define DEVINL __device__ __forceinline__

// ---------------- problem constants ----------------
constexpr int E    = 256;
constexpr int H    = 256;
constexpr int KT   = 16;     // CRF tags
constexpr int HD   = 128;
constexpr int B    = 32;
constexpr int NSEG = 8;
constexpr int S    = 128;
constexpr int T    = NSEG * S;     // 1024
constexpr int G4   = 4 * HD;       // 512
constexpr int MROWS = T * B;       // 32768 rows for the big GEMMs

// bf16 weight pool element offsets (ushort units)
constexpr int WIHF = 0;
constexpr int WHHF = WIHF + G4 * E;        // 131072
constexpr int WIHB = WHHF + G4 * HD;       // 196608
constexpr int WHHB = WIHB + G4 * E;        // 327680
constexpr int WLIN = WHHB + G4 * HD;       // 393216
constexpr int WTOT = WLIN + KT * H;        // 397312

// ---------------- workspace layout (bytes) ----------------
constexpr size_t OFF_XF32 = 0;                                         // f32 [NSEG,S,B,E]
constexpr size_t OFF_XBF  = OFF_XF32 + (size_t)MROWS * E * 4;          // bf16 same
constexpr size_t OFF_G1F  = OFF_XBF  + (size_t)MROWS * E * 2;          // bf16 [MROWS,512]
constexpr size_t OFF_G1B  = OFF_G1F  + (size_t)MROWS * G4 * 2;
constexpr size_t OFF_L1   = OFF_G1B  + (size_t)MROWS * G4 * 2;         // f32 [T,B,H]
constexpr size_t OFF_L2IN = OFF_L1   + (size_t)MROWS * H * 4;          // bf16 [T,B,H]
constexpr size_t OFF_G2F  = OFF_L2IN + (size_t)MROWS * H * 2;
constexpr size_t OFF_G2B  = OFF_G2F  + (size_t)MROWS * G4 * 2;
constexpr size_t OFF_L2   = OFF_G2B  + (size_t)MROWS * G4 * 2;         // f32 [T,B,H]
constexpr size_t OFF_EMIS = OFF_L2   + (size_t)MROWS * H * 4;          // f32 [T,B,16]
constexpr size_t OFF_HIST = OFF_EMIS + (size_t)MROWS * KT * 4;         // i32 [T-1,B,16]
constexpr size_t OFF_PERM = OFF_HIST + (size_t)(T - 1) * B * KT * 4;   // i32 [B,T]
constexpr size_t OFF_NEWL = OFF_PERM + (size_t)B * T * 4;              // i32 [B]
constexpr size_t OFF_HID  = OFF_NEWL + 256;                            // f32 [2][2][B][HD]
constexpr size_t OFF_WBF  = OFF_HID  + (size_t)4 * B * HD * 4;         // bf16 weight pool
constexpr size_t OFF_BSUM = OFF_WBF  + (size_t)WTOT * 2;               // f32 [2][512]

// dynamic LDS for the recurrent kernel (Whh now lives in VGPRs)
constexpr int LSTM_LDS = B * HD * 2      /* h bf16      8192 */
                       + B * HD * 4      /* h f32      16384 */
                       + B * HD * 4      /* c f32      16384 */
                       + B * G4 * 4      /* gate buf   65536 */
                       + 128;            /* lens            */

// ---------------- WMMA plumbing ----------------
typedef __attribute__((ext_vector_type(16))) __bf16          v16bf;
typedef __attribute__((ext_vector_type(16))) unsigned short  v16us;
typedef __attribute__((ext_vector_type(8)))  float           v8f;

union FragBF { v16bf b; v16us u; unsigned int d[8]; uint4 q[2]; };

DEVINL v8f vzero8() { v8f z = {0.f,0.f,0.f,0.f,0.f,0.f,0.f,0.f}; return z; }

DEVINL unsigned short f2bf(float f) {
  unsigned int u = __float_as_uint(f);
  unsigned int r = (u + 0x7FFFu + ((u >> 16) & 1u)) >> 16;
  return (unsigned short)r;
}
DEVINL float bf2f(unsigned short h) { return __uint_as_float(((unsigned int)h) << 16); }

DEVINL v8f wmma_bf16(const FragBF& a, const FragBF& b, v8f c) {
  return __builtin_amdgcn_wmma_f32_16x16x32_bf16(false, a.b, false, b.b, (short)0, c, false, false);
}

// A fragment (16x32, bf16): lane holds row m=lane&15; dword j holds K pair
// kb(j) = half*16 + 2*j   (contiguous, matches B fragment)
DEVINL void load_a_frag(FragBF& f, const unsigned short* row, int half) {
#pragma unroll
  for (int j = 0; j < 8; j++) {
    int kb = half * 16 + 2 * j;
    f.d[j] = *(const unsigned int*)(row + kb);
  }
}
DEVINL int a_elem_k(int e, int half) {   // scalar-fill variant (for f32->bf16 on the fly)
  int j = e >> 1, p = e & 1;
  return half * 16 + 2 * j + p;
}
// B fragment (32x16, bf16): lane holds col n=lane&15, K = half*16 + e, contiguous
DEVINL void load_b_frag(FragBF& f, const unsigned short* p) {
  f.q[0] = *(const uint4*)p;
  f.q[1] = *(const uint4*)(p + 8);
}

DEVINL float sigf(float x) { return 1.f / (1.f + expf(-x)); }

// ---------------- kernel 1: weight conversion + bias fusion ----------------
__global__ __launch_bounds__(256) void prep_kernel(
    const float* __restrict__ Wih_f, const float* __restrict__ Whh_f,
    const float* __restrict__ Wih_b, const float* __restrict__ Whh_b,
    const float* __restrict__ Wlin,
    const float* __restrict__ bih_f, const float* __restrict__ bhh_f,
    const float* __restrict__ bih_b, const float* __restrict__ bhh_b,
    unsigned short* __restrict__ wbf, float* __restrict__ bsum) {
  int idx = blockIdx.x * 256 + threadIdx.x;
  if (idx < WHHF)            wbf[idx] = f2bf(Wih_f[idx - WIHF]);
  else if (idx < WIHB)       wbf[idx] = f2bf(Whh_f[idx - WHHF]);
  else if (idx < WHHB)       wbf[idx] = f2bf(Wih_b[idx - WIHB]);
  else if (idx < WLIN)       wbf[idx] = f2bf(Whh_b[idx - WHHB]);
  else if (idx < WTOT)       wbf[idx] = f2bf(Wlin[idx - WLIN]);
  else if (idx < WTOT + G4)  { int j = idx - WTOT;      bsum[j]      = bih_f[j] + bhh_f[j]; }
  else if (idx < WTOT + 2*G4){ int j = idx - WTOT - G4; bsum[G4 + j] = bih_b[j] + bhh_b[j]; }
}

// ---------------- kernel 2: embedding gather -> x[seg,s,b,e] ----------------
__global__ __launch_bounds__(256) void embed_kernel(
    const int* __restrict__ texts, const float* __restrict__ emb,
    float* __restrict__ xf, unsigned short* __restrict__ xb) {
  int row = blockIdx.x;                 // (seg*S+s)*B + b
  int b   = row & (B - 1);
  int ts  = row >> 5;
  int s   = ts & (S - 1);
  int seg = ts >> 7;
  int tok = texts[(b * NSEG + seg) * S + s];
  int e = threadIdx.x;
  float v = emb[(size_t)tok * E + e];
  xf[(size_t)row * E + e] = v;
  xb[(size_t)row * E + e] = f2bf(v);
}

// ---------------- kernel 3: gates GEMM  C[m,n]=sum_k A[m,k]*W[n,k]+bias[n] ----
// M=32768, K=256, N=512, bf16 in, bf16 out. Block tile 128x64, wave tile 32x32.
__global__ __launch_bounds__(256) void gates_gemm_kernel(
    const unsigned short* __restrict__ A, const unsigned short* __restrict__ W,
    const float* __restrict__ bias, unsigned short* __restrict__ C) {
  __shared__ unsigned short As[128 * 32];
  __shared__ unsigned short Ws[64 * 32];
  const int tid  = threadIdx.x;
  const int lane = tid & 31, wid = tid >> 5, half = lane >> 4;
  const int wm = wid >> 1, wn = wid & 1;
  const int bm = blockIdx.x >> 3, bn = blockIdx.x & 7;
  const int m0 = bm * 128, n0 = bn * 64;

  v8f acc[2][2];
#pragma unroll
  for (int i = 0; i < 2; i++)
#pragma unroll
    for (int j = 0; j < 2; j++) acc[i][j] = vzero8();

  for (int kk = 0; kk < 256; kk += 32) {
    // stage A (128x32) and W (64x32) tiles through LDS, 128-bit chunks
    {
      const uint4* asrc = (const uint4*)A;
      uint4* adst = (uint4*)As;
#pragma unroll
      for (int j = 0; j < 2; j++) {
        int e8 = tid * 2 + j;                 // 512 chunks of 8 bf16
        int r = e8 >> 2, c8 = (e8 & 3) * 8;
        adst[e8] = asrc[((size_t)(m0 + r) * 256 + kk + c8) >> 3];
      }
      const uint4* wsrc = (const uint4*)W;
      uint4* wdst = (uint4*)Ws;
      int r = tid >> 2, c8 = (tid & 3) * 8;   // 256 chunks of 8 bf16
      wdst[tid] = wsrc[((size_t)(n0 + r) * 256 + kk + c8) >> 3];
      if (kk < 224)  // warm L2/WGP$ for next K slice
        __builtin_prefetch(&A[(size_t)(m0 + (tid >> 1)) * 256 + kk + 32], 0, 0);
    }
    __syncthreads();
    FragBF a[2], bf_[2];
#pragma unroll
    for (int mt = 0; mt < 2; mt++)
      load_a_frag(a[mt], &As[(wm * 32 + mt * 16 + (lane & 15)) * 32], half);
#pragma unroll
    for (int nt = 0; nt < 2; nt++)
      load_b_frag(bf_[nt], &Ws[(wn * 32 + nt * 16 + (lane & 15)) * 32 + half * 16]);
#pragma unroll
    for (int mt = 0; mt < 2; mt++)
#pragma unroll
      for (int nt = 0; nt < 2; nt++)
        acc[mt][nt] = wmma_bf16(a[mt], bf_[nt], acc[mt][nt]);
    __syncthreads();
  }
#pragma unroll
  for (int mt = 0; mt < 2; mt++)
#pragma unroll
    for (int nt = 0; nt < 2; nt++)
#pragma unroll
      for (int r = 0; r < 8; r++) {
        int m = m0 + wm * 32 + mt * 16 + r + 8 * half;  // C layout: VGPR r -> M=r / r+8
        int n = n0 + wn * 32 + nt * 16 + (lane & 15);
        C[(size_t)m * G4 + n] = f2bf(acc[mt][nt][r] + bias[n]);
      }
}

// ---------------- kernel 4: persistent recurrent LSTM scan ----------------
// 512 threads = 16 waves; each wave owns 32 gate columns and keeps its whole
// Whh slice (8 B-fragments = 64 VGPRs) register-resident across the time loop.
// mode 0: 16 blocks = (seg,dir), 128 steps each. mode 1: 2 blocks = dir, 1024 steps.
__global__ __launch_bounds__(512) void lstm_kernel(
    const unsigned short* __restrict__ wbf,
    const unsigned short* __restrict__ gF, const unsigned short* __restrict__ gB,
    const int* __restrict__ lens, float* __restrict__ out,
    const float* __restrict__ init, float* __restrict__ hidout, int mode) {
  extern __shared__ char smem[];
  unsigned short* hbf = (unsigned short*)smem;             // [32][128] bf16
  float* hf   = (float*)(hbf + B * HD);                    // [32][128]
  float* cf   = hf + B * HD;                               // [32][128]
  float* gbuf = cf + B * HD;                               // [32][512]
  int*   lenS = (int*)(gbuf + B * G4);

  int dir, seg, steps, tbase;
  if (mode == 0) { seg = blockIdx.x >> 1; dir = blockIdx.x & 1; steps = S; tbase = seg * S; }
  else           { seg = 0; dir = blockIdx.x; steps = T; tbase = 0; }

  const unsigned short* whh   = wbf + (dir ? WHHB : WHHF);   // bf16 [512][128]
  const unsigned short* gates = dir ? gB : gF;
  const int tid = threadIdx.x;
  const int lane = tid & 31, wid = tid >> 5, half = lane >> 4;
  const int n0 = wid * 32;   // each of 16 waves owns 32 gate columns

  // hoist Whh into registers: bw[kt][nt], kt = K/32 tile, nt = N/16 tile
  FragBF bw[4][2];
#pragma unroll
  for (int kt = 0; kt < 4; kt++)
#pragma unroll
    for (int nt = 0; nt < 2; nt++)
      load_b_frag(bw[kt][nt],
                  whh + (size_t)(n0 + nt * 16 + (lane & 15)) * HD + kt * 32 + half * 16);

  for (int i = tid; i < B * HD; i += 512) {
    float h0 = 0.f, c0 = 0.f;
    if (mode == 1) {
      h0 = init[(size_t)dir * 2 * B * HD + i];
      c0 = init[(size_t)dir * 2 * B * HD + B * HD + i];
    }
    hf[i] = h0; cf[i] = c0; hbf[i] = f2bf(h0);
  }
  if (tid < B) lenS[tid] = (mode == 0) ? lens[tid * NSEG + seg] : lens[tid];
  __syncthreads();

  for (int s = 0; s < steps; s++) {
    const int sid = dir ? (steps - 1 - s) : s;
    const int tg  = tbase + sid;

    // --- Wh = h @ Whh^T  (32x128 x 128x512): 16 WMMA per wave, B from VGPRs ---
    v8f acc[2][2];
#pragma unroll
    for (int i = 0; i < 2; i++)
#pragma unroll
      for (int j = 0; j < 2; j++) acc[i][j] = vzero8();
#pragma unroll
    for (int kt = 0; kt < 4; kt++) {
      const int kk = kt * 32;
      FragBF a[2];
#pragma unroll
      for (int mt = 0; mt < 2; mt++)
        load_a_frag(a[mt], &hbf[(mt * 16 + (lane & 15)) * HD + kk], half);
#pragma unroll
      for (int nt = 0; nt < 2; nt++) {
        acc[0][nt] = wmma_bf16(a[0], bw[kt][nt], acc[0][nt]);
        acc[1][nt] = wmma_bf16(a[1], bw[kt][nt], acc[1][nt]);
      }
    }
#pragma unroll
    for (int mt = 0; mt < 2; mt++)
#pragma unroll
      for (int nt = 0; nt < 2; nt++)
#pragma unroll
        for (int r = 0; r < 8; r++) {
          int m = mt * 16 + r + 8 * half;     // C layout: VGPR r -> M=r / r+8
          int n = n0 + nt * 16 + (lane & 15);
          gbuf[m * G4 + n] = acc[mt][nt][r];
        }
    __syncthreads();

    // --- gate nonlinearities + state update; thread -> (b, 8 d's) ---
    {
      const int b  = tid >> 4;
      const int d0 = (tid & 15) * 8;
      const unsigned short* grow = gates + ((size_t)tg * B + b) * G4;
      const bool msk = sid < lenS[b];
      float* orow = out + ((size_t)tg * B + b) * H + dir * HD;
#pragma unroll 4
      for (int j = 0; j < 8; j++) {
        int d = d0 + j;
        float ig = gbuf[b * G4 + d]           + bf2f(grow[d]);
        float fg = gbuf[b * G4 + HD + d]      + bf2f(grow[HD + d]);
        float gg = gbuf[b * G4 + 2 * HD + d]  + bf2f(grow[2 * HD + d]);
        float og = gbuf[b * G4 + 3 * HD + d]  + bf2f(grow[3 * HD + d]);
        float cn = sigf(fg) * cf[b * HD + d] + sigf(ig) * tanhf(gg);
        float hn = sigf(og) * tanhf(cn);
        float hnew = msk ? hn : hf[b * HD + d];
        float cnew = msk ? cn : cf[b * HD + d];
        hf[b * HD + d] = hnew;
        cf[b * HD + d] = cnew;
        hbf[b * HD + d] = f2bf(hnew);
        orow[d] = msk ? hn : 0.f;
      }
    }
    __syncthreads();
  }

  if (mode == 0 && seg == NSEG - 1 && hidout) {   // export carries for stage 2 init
    float* hd = hidout + (size_t)dir * 2 * B * HD;
    for (int i = tid; i < B * HD; i += 512) { hd[i] = hf[i]; hd[B * HD + i] = cf[i]; }
  }
}

// ---------------- kernel 5: replace rows s >= maxlen(seg) with x ----------------
__global__ __launch_bounds__(256) void overfix_kernel(
    const int* __restrict__ lengths, const float* __restrict__ xf, float* __restrict__ l1) {
  int row = blockIdx.x;           // seg*S + s
  int seg = row >> 7, s = row & (S - 1);
  int maxlen = 0;
  for (int b = 0; b < B; b++) maxlen = max(maxlen, lengths[b * NSEG + seg]);
  if (s < maxlen) return;
  size_t base = (size_t)row * B * H;
  for (int i = threadIdx.x; i < B * H; i += blockDim.x) l1[base + i] = xf[base + i];
}

// ---------------- kernel 6: stable pack permutation + new lengths ----------------
__global__ void perm_kernel(const int* __restrict__ lengths,
                            int* __restrict__ perm, int* __restrict__ newlen) {
  int b = threadIdx.x;
  if (b >= B) return;
  int nl = 0;
  for (int seg = 0; seg < NSEG; seg++) nl += lengths[b * NSEG + seg];
  newlen[b] = nl;
  int pv = 0, pi = nl;
  for (int t = 0; t < T; t++) {
    int seg = t >> 7, pos = t & (S - 1);
    if (pos < lengths[b * NSEG + seg]) perm[b * T + (pv++)] = t;
    else                               perm[b * T + (pi++)] = t;
  }
}

// ---------------- kernel 7: gather lstm2 input (f32 -> bf16) ----------------
__global__ __launch_bounds__(256) void gather_kernel(
    const int* __restrict__ perm, const float* __restrict__ l1,
    unsigned short* __restrict__ l2in) {
  int row = blockIdx.x;           // t*B + b
  int b = row & (B - 1), t = row >> 5;
  int src = perm[b * T + t];
  const float* sp = l1 + ((size_t)src * B + b) * H;
  unsigned short* dp = l2in + (size_t)row * H;
  int i = threadIdx.x;
  dp[i] = f2bf(sp[i]);
}

// ---------------- kernel 8: emission GEMM  [32768x256] x [256x16] ----------------
__global__ __launch_bounds__(256) void emis_kernel(
    const float* __restrict__ l2, const unsigned short* __restrict__ wlin,
    const float* __restrict__ blin, float* __restrict__ emis) {
  __shared__ unsigned short wl[KT * H];
  const int tid = threadIdx.x, lane = tid & 31, wid = tid >> 5, half = lane >> 4;
  for (int i = tid; i < KT * H / 8; i += 256) ((uint4*)wl)[i] = ((const uint4*)wlin)[i];
  __syncthreads();
  const int m0 = blockIdx.x * 128 + wid * 16;
  v8f acc = vzero8();
  for (int kk = 0; kk < H; kk += 32) {
    FragBF a, bf_;
    int mrow = m0 + (lane & 15);
#pragma unroll
    for (int e = 0; e < 16; e++)
      a.u[e] = f2bf(l2[(size_t)mrow * H + kk + a_elem_k(e, half)]);
    load_b_frag(bf_, &wl[(lane & 15) * H + kk + half * 16]);
    acc = wmma_bf16(a, bf_, acc);
  }
#pragma unroll
  for (int r = 0; r < 8; r++) {
    int m = m0 + r + 8 * half;
    int n = lane & 15;
    emis[(size_t)m * KT + n] = acc[r] + blin[n];
  }
}

// ---------------- kernel 9: Viterbi (forward + backtrace) ----------------
__global__ __launch_bounds__(512) void viterbi_kernel(
    const float* __restrict__ emis, const float* __restrict__ start,
    const float* __restrict__ trans, const float* __restrict__ endv,
    const int* __restrict__ newlen, int* __restrict__ hist, float* __restrict__ outv) {
  __shared__ float sc[B][KT];
  __shared__ float tr[KT][KT];
  __shared__ float fin[B][KT];
  __shared__ int   nl[B];
  __shared__ int   last[B];
  const int tid = threadIdx.x;
  const int b = tid >> 4, k = tid & 15;
  if (tid < KT * KT) ((float*)tr)[tid] = trans[tid];
  if (tid < B) nl[tid] = newlen[tid];
  __syncthreads();
  sc[b][k] = start[k] + emis[(size_t)b * KT + k];
  __syncthreads();
  for (int t = 1; t < T; t++) {
    float em = emis[((size_t)t * B + b) * KT + k];
    float best = -3.4e38f; int bi = 0;
#pragma unroll
    for (int kp = 0; kp < KT; kp++) {
      float v = sc[b][kp] + tr[kp][k];
      if (v > best) { best = v; bi = kp; }
    }
    hist[((size_t)(t - 1) * B + b) * KT + k] = bi;
    float ns = (t < nl[b]) ? (best + em) : sc[b][k];
    __syncthreads();
    sc[b][k] = ns;
    __syncthreads();
  }
  fin[b][k] = sc[b][k] + endv[k];
  __syncthreads();
  if (k == 0) {
    float best = fin[b][0]; int bi = 0;
    for (int kk = 1; kk < KT; kk++) if (fin[b][kk] > best) { best = fin[b][kk]; bi = kk; }
    last[b] = bi;
    outv[(size_t)T * B + b] = best;     // best_score
  }
  __syncthreads();
  if (tid < B) {
    int bb = tid;
    int cur = last[bb];
    outv[(size_t)(T - 1) * B + bb] = (float)cur;
    for (int t = T - 2; t >= 0; t--) {
      int prev = hist[((size_t)t * B + bb) * KT + cur];
      if ((t + 1) < nl[bb]) cur = prev;
      outv[(size_t)t * B + bb] = (float)cur;
    }
  }
}

// ---------------- host launcher ----------------
extern "C" void kernel_launch(void* const* d_in, const int* in_sizes, int n_in,
                              void* d_out, int out_size, void* d_ws, size_t ws_size,
                              hipStream_t stream) {
  (void)in_sizes; (void)n_in; (void)out_size; (void)ws_size;
  const int*   texts   = (const int*)  d_in[0];
  const int*   lengths = (const int*)  d_in[1];
  const float* emb     = (const float*)d_in[2];
  const float* Wih_f   = (const float*)d_in[3];
  const float* Whh_f   = (const float*)d_in[4];
  const float* bih_f   = (const float*)d_in[5];
  const float* bhh_f   = (const float*)d_in[6];
  const float* Wih_b   = (const float*)d_in[7];
  const float* Whh_b   = (const float*)d_in[8];
  const float* bih_b   = (const float*)d_in[9];
  const float* bhh_b   = (const float*)d_in[10];
  const float* Wlin    = (const float*)d_in[11];
  const float* blin    = (const float*)d_in[12];
  const float* crf_s   = (const float*)d_in[13];
  const float* crf_t   = (const float*)d_in[14];
  const float* crf_e   = (const float*)d_in[15];

  char* ws = (char*)d_ws;
  float*          XF   = (float*)         (ws + OFF_XF32);
  unsigned short* XB   = (unsigned short*)(ws + OFF_XBF);
  unsigned short* G1F  = (unsigned short*)(ws + OFF_G1F);
  unsigned short* G1B  = (unsigned short*)(ws + OFF_G1B);
  float*          L1   = (float*)         (ws + OFF_L1);
  unsigned short* L2IN = (unsigned short*)(ws + OFF_L2IN);
  unsigned short* G2F  = (unsigned short*)(ws + OFF_G2F);
  unsigned short* G2B  = (unsigned short*)(ws + OFF_G2B);
  float*          L2   = (float*)         (ws + OFF_L2);
  float*          EMIS = (float*)         (ws + OFF_EMIS);
  int*            HIST = (int*)           (ws + OFF_HIST);
  int*            PERM = (int*)           (ws + OFF_PERM);
  int*            NEWL = (int*)           (ws + OFF_NEWL);
  float*          HID  = (float*)         (ws + OFF_HID);
  unsigned short* WBF  = (unsigned short*)(ws + OFF_WBF);
  float*          BSUM = (float*)         (ws + OFF_BSUM);
  float*          OUT  = (float*)d_out;

  (void)hipFuncSetAttribute((const void*)lstm_kernel,
                            hipFuncAttributeMaxDynamicSharedMemorySize, LSTM_LDS);

  // 1. weights -> bf16, fused biases
  prep_kernel<<<(WTOT + 2 * G4 + 255) / 256, 256, 0, stream>>>(
      Wih_f, Whh_f, Wih_b, Whh_b, Wlin, bih_f, bhh_f, bih_b, bhh_b, WBF, BSUM);
  // 2. embedding gather
  embed_kernel<<<MROWS, E, 0, stream>>>(texts, emb, XF, XB);
  // 3. stage-1 input-gate GEMMs (both directions)
  gates_gemm_kernel<<<(MROWS / 128) * (G4 / 64), 256, 0, stream>>>(XB, WBF + WIHF, BSUM,      G1F);
  gates_gemm_kernel<<<(MROWS / 128) * (G4 / 64), 256, 0, stream>>>(XB, WBF + WIHB, BSUM + G4, G1B);
  // 4. stage-1 recurrent scan: 16 persistent blocks (seg x dir)
  lstm_kernel<<<NSEG * 2, 512, LSTM_LDS, stream>>>(WBF, G1F, G1B, lengths, L1,
                                                   nullptr, HID, 0);
  // 5. over-length rows get the raw embedding back
  overfix_kernel<<<NSEG * S, 256, 0, stream>>>(lengths, XF, L1);
  // 6. pack permutation + total lengths
  perm_kernel<<<1, 32, 0, stream>>>(lengths, PERM, NEWL);
  // 7. gather packed sequence, convert to bf16
  gather_kernel<<<MROWS, H, 0, stream>>>(PERM, L1, L2IN);
  // 8. stage-2 input-gate GEMMs
  gates_gemm_kernel<<<(MROWS / 128) * (G4 / 64), 256, 0, stream>>>(L2IN, WBF + WIHF, BSUM,      G2F);
  gates_gemm_kernel<<<(MROWS / 128) * (G4 / 64), 256, 0, stream>>>(L2IN, WBF + WIHB, BSUM + G4, G2B);
  // 9. stage-2 recurrent scan over full T with stage-1 carries
  lstm_kernel<<<2, 512, LSTM_LDS, stream>>>(WBF, G2F, G2B, NEWL, L2, HID, nullptr, 1);
  // 10. emission projection (WMMA, N=16)
  emis_kernel<<<MROWS / 128, 256, 0, stream>>>(L2, WBF + WLIN, blin, EMIS);
  // 11. Viterbi decode -> tags (as float) + best_score
  viterbi_kernel<<<1, 512, 0, stream>>>(EMIS, crf_s, crf_t, crf_e, NEWL, HIST, OUT);
}
